// TriangleMultiplication_23124103922385
// MI455X (gfx1250) — matrix operand order repaired and checked by reading, not compiled, not run on hardware
//
#include <hip/hip_runtime.h>

// ---------------------------------------------------------------------------
// Triangle multiplication (incoming), B=1, N=512, DIM=DIM_H=128
//   k0: convert 6 weight matrices f32 -> f16
//   k1: LayerNorm(x) + 5 projections (WMMA f16) + gates; left/right stored in
//       [h][row][k] transposed f16 layout + ogate f16
//   k2: per-channel 512x512x512 NT GEMM (WMMA f16), double-buffered async
//       global->LDS staging (ASYNCcnt), 128 channels -> tri f32
//   k3: LayerNorm(tri) * sigmoid(ogate), project W_out (WMMA f16) -> out f32
// ---------------------------------------------------------------------------

typedef __attribute__((ext_vector_type(16))) _Float16 v16h;
typedef __attribute__((ext_vector_type(8)))  _Float16 v8h;
typedef __attribute__((ext_vector_type(8)))  float    v8f;
typedef int v4i __attribute__((vector_size(16)));   // matches async-LDS builtin

union Frag16 { v16h v; v8h h[2]; };

#define NN 512
#define DD 128

#if defined(__gfx1250__) &&                                                    \
    __has_builtin(__builtin_amdgcn_global_load_async_to_lds_b128) &&           \
    __has_builtin(__builtin_amdgcn_s_wait_asynccnt)
#define HAVE_ASYNC_LDS 1
#define GAS(p) ((__attribute__((address_space(1))) v4i*)(p))
#define LAS(p) ((__attribute__((address_space(3))) v4i*)(p))
#endif

__device__ __forceinline__ float sigf(float x) { return 1.0f / (1.0f + __expf(-x)); }

__device__ __forceinline__ v8f wmma16(v16h a, v16h b, v8f c) {
    // D = A(16x32 f16) * B(32x16 f16) + C(16x16 f32)
    return __builtin_amdgcn_wmma_f32_16x16x32_f16(false, a, false, b, (short)0, c,
                                                  false, false);
}

// --------------------------- k0: weight convert ----------------------------
__global__ __launch_bounds__(256) void k0_cvt(const float* __restrict__ w0,
                                              const float* __restrict__ w1,
                                              const float* __restrict__ w2,
                                              const float* __restrict__ w3,
                                              const float* __restrict__ w4,
                                              const float* __restrict__ w5,
                                              _Float16* __restrict__ dst) {
    int i = blockIdx.x * 256 + threadIdx.x;     // 6*128*128 = 98304 total
    int m = i >> 14, off = i & 16383;
    const float* s;
    switch (m) {
        case 0: s = w0; break;
        case 1: s = w1; break;
        case 2: s = w2; break;
        case 3: s = w3; break;
        case 4: s = w4; break;
        default: s = w5; break;
    }
    dst[i] = (_Float16)s[off];
}

// ------------------- k1: LN + 5 projections + gating -----------------------
__global__ __launch_bounds__(256) void k1_ln_proj(
    const float* __restrict__ x, const unsigned char* __restrict__ mask,
    const float* __restrict__ ln_g, const float* __restrict__ ln_b,
    const _Float16* __restrict__ wf16,          // [5][128][128] f16
    _Float16* __restrict__ leftT,               // [h][q][p] f16
    _Float16* __restrict__ rightT,              // [h][q][p] f16
    _Float16* __restrict__ ogbuf)               // [p][q][h] f16
{
    __shared__ _Float16 xn[32 * 136];           // padded stride: no bank conflicts
    const int tid = threadIdx.x, wave = tid >> 5, lane = tid & 31;
    const int R0 = blockIdx.x * 32;             // global pair-row base
    const int p = R0 >> 9, q0 = R0 & 511;

    // ---- LayerNorm over D=128: each wave handles 4 rows, 32 lanes/row ----
    for (int sub = 0; sub < 4; ++sub) {
        const int row = wave * 4 + sub;
        const float4 xv = *(const float4*)(x + (size_t)(R0 + row) * DD + lane * 4);
        float s  = xv.x + xv.y + xv.z + xv.w;
        float s2 = xv.x * xv.x + xv.y * xv.y + xv.z * xv.z + xv.w * xv.w;
        #pragma unroll
        for (int off = 16; off > 0; off >>= 1) {
            s  += __shfl_xor(s, off);
            s2 += __shfl_xor(s2, off);
        }
        const float mu = s * (1.0f / 128.0f);
        const float var = s2 * (1.0f / 128.0f) - mu * mu;
        const float rstd = rsqrtf(var + 1e-5f);
        const float vals[4] = {xv.x, xv.y, xv.z, xv.w};
        #pragma unroll
        for (int t = 0; t < 4; ++t) {
            const int d = lane * 4 + t;
            xn[row * 136 + d] = (_Float16)((vals[t] - mu) * rstd * ln_g[d] + ln_b[d]);
        }
    }
    __syncthreads();

    const int lane16 = lane & 15, hi = lane >> 4;
    const unsigned char maskp = mask[p];

    for (int tt = 0; tt < 2; ++tt) {
        const int t = wave * 2 + tt;
        const int mt = t >> 3, c = t & 7;
        const int hcol = c * 16 + lane16;
        v8f acc[5];
        #pragma unroll
        for (int w5 = 0; w5 < 5; ++w5) acc[w5] = (v8f)(0.0f);

        #pragma unroll
        for (int kc = 0; kc < 4; ++kc) {
            Frag16 a;
            const int abase = (mt * 16 + lane16) * 136 + kc * 32 + hi * 16;
            a.h[0] = *(const v8h*)(xn + abase);
            a.h[1] = *(const v8h*)(xn + abase + 8);
            #pragma unroll
            for (int w5 = 0; w5 < 5; ++w5) {
                Frag16 b;
                const _Float16* wp =
                    wf16 + (((size_t)w5 * 128 + hcol) << 7) + kc * 32 + hi * 16;
                b.h[0] = *(const v8h*)wp;
                b.h[1] = *(const v8h*)(wp + 8);
                acc[w5] = wmma16(a.v, b.v, acc[w5]);
            }
        }

        #pragma unroll
        for (int r = 0; r < 8; ++r) {
            const int mrow = mt * 16 + r + hi * 8;
            const int q = q0 + mrow;
            const float mf = (maskp && mask[q]) ? 1.0f : 0.0f;
            const float lv = acc[0][r] * mf * sigf(acc[2][r]);
            const float rv = acc[1][r] * mf * sigf(acc[3][r]);
            const float og = sigf(acc[4][r]);
            const size_t ti = ((size_t)hcol * NN + q) * NN + p;
            leftT[ti]  = (_Float16)lv;
            rightT[ti] = (_Float16)rv;
            ogbuf[((size_t)(p * NN + q)) * DD + hcol] = (_Float16)og;
        }
    }
}

// ----------------- k2: 128 x (512x512x512) triangle GEMMs ------------------
// out[i,j,h] = sum_k right[k,i,h] * left[k,j,h]. With the [h][row][k] layout
// this is an NT GEMM: A rows and B columns both contiguous over k.
// Double-buffered: chunk kc+1 streams global->LDS via async-DMA while WMMAs
// consume chunk kc.
__global__ __launch_bounds__(256) void k2_tri(const _Float16* __restrict__ leftT,
                                              const _Float16* __restrict__ rightT,
                                              float* __restrict__ tri) {
    __shared__ _Float16 As[2][64 * 40];         // 64 i-rows x 32 k (padded)
    __shared__ _Float16 Bs[2][64 * 40];         // 64 j-rows x 32 k (padded)
    const int tid = threadIdx.x, wave = tid >> 5, lane = tid & 31;
    const int h = blockIdx.z, i0 = blockIdx.y * 64, j0 = blockIdx.x * 64;
    const int lane16 = lane & 15, hi = lane >> 4;
    const int mi = wave >> 1, nj0 = (wave & 1) * 2;  // wave -> one i-tile, two j-tiles
    const size_t hbase = (size_t)h * NN * NN;

    const int lr = tid >> 2, seg = (tid & 3) * 8;    // per-thread 16B segment
    const _Float16* srcA = rightT + hbase + (size_t)(i0 + lr) * NN + seg;
    const _Float16* srcB = leftT  + hbase + (size_t)(j0 + lr) * NN + seg;
    const int ldst = lr * 40 + seg;

    v8f acc0 = (v8f)(0.0f), acc1 = (v8f)(0.0f);

#if defined(HAVE_ASYNC_LDS)
    // prologue: stream chunk 0 into buffer 0
    __builtin_amdgcn_global_load_async_to_lds_b128(GAS(srcA), LAS(&As[0][ldst]), 0, 0);
    __builtin_amdgcn_global_load_async_to_lds_b128(GAS(srcB), LAS(&Bs[0][ldst]), 0, 0);

    for (int kc = 0; kc < 16; ++kc) {
        const int cur = kc & 1;
        if (kc < 15) {
            // stream chunk kc+1 into the other buffer while we compute on cur
            __builtin_amdgcn_global_load_async_to_lds_b128(
                GAS(srcA + (kc + 1) * 32), LAS(&As[cur ^ 1][ldst]), 0, 0);
            __builtin_amdgcn_global_load_async_to_lds_b128(
                GAS(srcB + (kc + 1) * 32), LAS(&Bs[cur ^ 1][ldst]), 0, 0);
            __builtin_amdgcn_s_wait_asynccnt(2);   // chunk-kc loads retired
        } else {
            __builtin_amdgcn_s_wait_asynccnt(0);
        }
        __syncthreads();                            // all waves' chunk-kc data in LDS

        Frag16 a, b0, b1;
        const int abase = (mi * 16 + lane16) * 40 + hi * 16;
        a.h[0] = *(const v8h*)(&As[cur][abase]);
        a.h[1] = *(const v8h*)(&As[cur][abase + 8]);
        const int bb0 = (nj0 * 16 + lane16) * 40 + hi * 16;
        b0.h[0] = *(const v8h*)(&Bs[cur][bb0]);
        b0.h[1] = *(const v8h*)(&Bs[cur][bb0 + 8]);
        const int bb1 = ((nj0 + 1) * 16 + lane16) * 40 + hi * 16;
        b1.h[0] = *(const v8h*)(&Bs[cur][bb1]);
        b1.h[1] = *(const v8h*)(&Bs[cur][bb1 + 8]);
        acc0 = wmma16(a.v, b0.v, acc0);
        acc1 = wmma16(a.v, b1.v, acc1);
        __syncthreads();                            // buffer reusable at kc+2
    }
#else
    for (int kc = 0; kc < 16; ++kc) {
        const int k0 = kc * 32;
        __syncthreads();
        *(v8h*)(&As[0][ldst]) = *(const v8h*)(srcA + k0);
        *(v8h*)(&Bs[0][ldst]) = *(const v8h*)(srcB + k0);
        __syncthreads();

        Frag16 a, b0, b1;
        const int abase = (mi * 16 + lane16) * 40 + hi * 16;
        a.h[0] = *(const v8h*)(&As[0][abase]);
        a.h[1] = *(const v8h*)(&As[0][abase + 8]);
        const int bb0 = (nj0 * 16 + lane16) * 40 + hi * 16;
        b0.h[0] = *(const v8h*)(&Bs[0][bb0]);
        b0.h[1] = *(const v8h*)(&Bs[0][bb0 + 8]);
        const int bb1 = ((nj0 + 1) * 16 + lane16) * 40 + hi * 16;
        b1.h[0] = *(const v8h*)(&Bs[0][bb1]);
        b1.h[1] = *(const v8h*)(&Bs[0][bb1 + 8]);
        acc0 = wmma16(a.v, b0.v, acc0);
        acc1 = wmma16(a.v, b1.v, acc1);
    }
#endif

    #pragma unroll
    for (int r = 0; r < 8; ++r) {
        const int i = i0 + mi * 16 + r + hi * 8;
        const int j = j0 + nj0 * 16 + lane16;
        tri[((size_t)i * NN + j) * DD + h]      = acc0[r];
        tri[((size_t)i * NN + j + 16) * DD + h] = acc1[r];
    }
}

// ------------- k3: LN(tri) * sigmoid(ogate), project with W_out ------------
__global__ __launch_bounds__(256) void k3_out(const float* __restrict__ tri,
                                              const _Float16* __restrict__ ogbuf,
                                              const float* __restrict__ on_g,
                                              const float* __restrict__ on_b,
                                              const _Float16* __restrict__ wout16,
                                              float* __restrict__ out) {
    __shared__ _Float16 an[32 * 136];
    const int tid = threadIdx.x, wave = tid >> 5, lane = tid & 31;
    const int R0 = blockIdx.x * 32;

    for (int sub = 0; sub < 4; ++sub) {
        const int row = wave * 4 + sub;
        const size_t R = (size_t)R0 + row;
        const float4 tv = *(const float4*)(tri + R * DD + lane * 4);
        float s  = tv.x + tv.y + tv.z + tv.w;
        float s2 = tv.x * tv.x + tv.y * tv.y + tv.z * tv.z + tv.w * tv.w;
        #pragma unroll
        for (int off = 16; off > 0; off >>= 1) {
            s  += __shfl_xor(s, off);
            s2 += __shfl_xor(s2, off);
        }
        const float mu = s * (1.0f / 128.0f);
        const float var = s2 * (1.0f / 128.0f) - mu * mu;
        const float rstd = rsqrtf(var + 1e-5f);
        const float vals[4] = {tv.x, tv.y, tv.z, tv.w};
        #pragma unroll
        for (int t = 0; t < 4; ++t) {
            const int d = lane * 4 + t;
            const float g = (float)ogbuf[R * DD + d];
            an[row * 136 + d] =
                (_Float16)(((vals[t] - mu) * rstd * on_g[d] + on_b[d]) * g);
        }
    }
    __syncthreads();

    const int lane16 = lane & 15, hi = lane >> 4;
    for (int tt = 0; tt < 2; ++tt) {
        const int t = wave * 2 + tt;
        const int mt = t >> 3, c = t & 7;
        const int dcol = c * 16 + lane16;
        v8f acc = (v8f)(0.0f);
        #pragma unroll
        for (int kc = 0; kc < 4; ++kc) {
            Frag16 a, b;
            const int abase = (mt * 16 + lane16) * 136 + kc * 32 + hi * 16;
            a.h[0] = *(const v8h*)(an + abase);
            a.h[1] = *(const v8h*)(an + abase + 8);
            const _Float16* wp = wout16 + ((size_t)dcol << 7) + kc * 32 + hi * 16;
            b.h[0] = *(const v8h*)wp;
            b.h[1] = *(const v8h*)(wp + 8);
            acc = wmma16(a.v, b.v, acc);
        }
        #pragma unroll
        for (int r = 0; r < 8; ++r) {
            const size_t R = (size_t)R0 + mt * 16 + r + hi * 8;
            out[R * DD + dcol] = acc[r];
        }
    }
}

// ---------------------------------------------------------------------------
extern "C" void kernel_launch(void* const* d_in, const int* in_sizes, int n_in,
                              void* d_out, int out_size, void* d_ws, size_t ws_size,
                              hipStream_t stream) {
    (void)in_sizes; (void)n_in; (void)out_size; (void)ws_size;
    const float*         x    = (const float*)d_in[0];
    const unsigned char* mask = (const unsigned char*)d_in[1];
    const float*         ln_g = (const float*)d_in[2];
    const float*         ln_b = (const float*)d_in[3];
    const float*         wl   = (const float*)d_in[4];
    const float*         wr   = (const float*)d_in[5];
    const float*         wlg  = (const float*)d_in[6];
    const float*         wrg  = (const float*)d_in[7];
    const float*         wog  = (const float*)d_in[8];
    const float*         on_g = (const float*)d_in[9];
    const float*         on_b = (const float*)d_in[10];
    const float*         wout = (const float*)d_in[11];
    float*               out  = (float*)d_out;

    char* ws = (char*)d_ws;
    size_t o = 0;
    _Float16* wf16   = (_Float16*)(ws + o); o += (size_t)6 * 128 * 128 * 2;   // 192 KB
    _Float16* leftT  = (_Float16*)(ws + o); o += (size_t)DD * NN * NN * 2;    // 64 MB
    _Float16* rightT = (_Float16*)(ws + o); o += (size_t)DD * NN * NN * 2;    // 64 MB
    _Float16* ogb    = (_Float16*)(ws + o); o += (size_t)NN * NN * DD * 2;    // 64 MB
    float*    tri    = (float*)(ws + o);                                      // 128 MB

    k0_cvt<<<384, 256, 0, stream>>>(wl, wr, wlg, wrg, wog, wout, wf16);
    k1_ln_proj<<<(NN * NN) / 32, 256, 0, stream>>>(x, mask, ln_g, ln_b, wf16,
                                                   leftT, rightT, ogb);
    k2_tri<<<dim3(NN / 64, NN / 64, DD), 256, 0, stream>>>(leftT, rightT, tri);
    k3_out<<<(NN * NN) / 32, 256, 0, stream>>>(tri, ogb, on_g, on_b,
                                               wf16 + (size_t)5 * 128 * 128, out);
}
